// PtrNet_14113262535167
// MI455X (gfx1250) — matrix operand order repaired and checked by reading, not compiled, not run on hardware
//
#include <hip/hip_runtime.h>
#include <hip/hip_bf16.h>

// Problem constants (from reference): B=64, L=128, T=32, H=512
#define B_  64
#define L_  128
#define T_  32
#define H_  512
#define G4H 2048   // 4*H

typedef __attribute__((ext_vector_type(8)))  __bf16 v8bf;
typedef __attribute__((ext_vector_type(16))) __bf16 v16bf;
typedef __attribute__((ext_vector_type(8)))  float  v8f;

// ---------------------------------------------------------------------------
// WMMA inner loop: computes one 16x16 f32 tile of  D = A(MxK) * W^T(KxN)
// A row-major [M x K] bf16 ; W row-major [N x K] bf16 (so B[k][n] = W[n][k]).
//
// CDNA5 16-bit A layout (16x32, MxK): lanes 0-15 = rows M0..15 holding
// K = k0+0..7 (elems 0..7) and k0+16..23 (elems 8..15); lanes 16-31 same rows
// holding K = k0+8..15 and k0+24..31.
// B layout (32x16, KxN): lane = N (mod 16), same K split as A: lanes 0-15
// hold K=k0+0..7,16..23; lanes 16-31 hold K=k0+8..15,24..31.
// C/D: lane<16 -> N=lane, M=elem; lane>=16 -> N=lane-16, M=elem+8.
// ---------------------------------------------------------------------------
__device__ __forceinline__ v8f wmma_bf16_tile(const __bf16* __restrict__ A,
                                              const __bf16* __restrict__ W,
                                              int Mbase, int Nbase, int K,
                                              v8f acc, int lane) {
  const int l15 = lane & 15;
  const int hi  = lane >> 4;                       // 0 or 1
  const __bf16* aRow = A + (size_t)(Mbase + l15) * K + (hi << 3);
  const __bf16* bRow = W + (size_t)(Nbase + l15) * K + (hi << 3);
  for (int k0 = 0; k0 < K; k0 += 32) {
    v8bf a_lo = *reinterpret_cast<const v8bf*>(aRow + k0);
    v8bf a_hi = *reinterpret_cast<const v8bf*>(aRow + k0 + 16);
    v8bf b_lo = *reinterpret_cast<const v8bf*>(bRow + k0);
    v8bf b_hi = *reinterpret_cast<const v8bf*>(bRow + k0 + 16);
    v16bf a, b;
#pragma unroll
    for (int e = 0; e < 8; ++e) {
      a[e] = a_lo[e]; a[8 + e] = a_hi[e];
      b[e] = b_lo[e]; b[8 + e] = b_hi[e];
    }
    // (neg_a, A, neg_b, B, c_mod, C, reuse_a, reuse_b)
    acc = __builtin_amdgcn_wmma_f32_16x16x32_bf16(false, a, false, b,
                                                  (short)0, acc, false, false);
  }
  return acc;
}

// ---------------------------------------------------------------------------
// Gates GEMM:  g[b, n] = sum_k h[b,k]*Whh[n,k] + bih[n]+bhh[n] + x[b]*Wih[n]
// M=64, N=2048, K=512. mode 0: x = seq[b, step]; mode 1 (decoder):
// x = target[b,0] (step 0), target[b,1] (step 1), else xnext[b].
// One 16x16 tile per wave; 512 tiles total -> 64 blocks x 256 threads.
// ---------------------------------------------------------------------------
__global__ void k_lstm_gates(const __bf16* __restrict__ hbf,
                             const __bf16* __restrict__ Whh,
                             const float* __restrict__ bih,
                             const float* __restrict__ bhh,
                             const float* __restrict__ wih,
                             const float* __restrict__ seq,
                             const float* __restrict__ target,
                             const float* __restrict__ xnext,
                             float* __restrict__ g,
                             int mode, int step) {
  const int wave = (blockIdx.x * blockDim.x + threadIdx.x) >> 5;
  const int lane = threadIdx.x & 31;
  const int NT = G4H / 16;                       // 128
  const int tiles = (B_ / 16) * NT;              // 512
  if (wave >= tiles) return;                     // wave-uniform guard
  const int mt = wave / NT, nt = wave % NT;
  const int Mbase = mt * 16, Nbase = nt * 16;
  const int l15 = lane & 15, hi = lane >> 4;
  const int n = Nbase + l15;

  const float bias = bih[n] + bhh[n];
  v8f acc;
#pragma unroll
  for (int r = 0; r < 8; ++r) acc[r] = bias;

  acc = wmma_bf16_tile(hbf, Whh, Mbase, Nbase, H_, acc, lane);

  const float wn = wih[n];
#pragma unroll
  for (int r = 0; r < 8; ++r) {
    const int m = Mbase + (hi << 3) + r;         // batch index
    float xv;
    if (mode == 0) {
      xv = seq[m * L_ + step];
    } else {
      xv = (step == 0) ? target[m * T_ + 0]
         : (step == 1) ? target[m * T_ + 1]
                       : xnext[m];
    }
    g[(size_t)m * G4H + n] = acc[r] + xv * wn;
  }
}

// ---------------------------------------------------------------------------
// Generic GEMM + bias:  out[m,n] = sum_k A[m,k]*W[n,k] + bias[n]
// Optional bf16 mirror output (for the iterated e-update ping-pong).
// ---------------------------------------------------------------------------
__global__ void k_gemm_bias(const __bf16* __restrict__ A,
                            const __bf16* __restrict__ W,
                            const float* __restrict__ bias,
                            float* __restrict__ out,
                            __bf16* __restrict__ out_bf,
                            int M, int N, int K) {
  const int wave = (blockIdx.x * blockDim.x + threadIdx.x) >> 5;
  const int lane = threadIdx.x & 31;
  const int NT = N / 16;
  const int tiles = (M / 16) * NT;
  if (wave >= tiles) return;                     // wave-uniform guard
  const int mt = wave / NT, nt = wave % NT;
  const int Mbase = mt * 16, Nbase = nt * 16;
  const int l15 = lane & 15, hi = lane >> 4;
  const int n = Nbase + l15;

  const float bn = bias[n];
  v8f acc;
#pragma unroll
  for (int r = 0; r < 8; ++r) acc[r] = bn;

  acc = wmma_bf16_tile(A, W, Mbase, Nbase, K, acc, lane);

#pragma unroll
  for (int r = 0; r < 8; ++r) {
    const int m = Mbase + (hi << 3) + r;
    const size_t o = (size_t)m * N + n;
    out[o] = acc[r];
    if (out_bf) out_bf[o] = (__bf16)acc[r];
  }
}

// ---------------------------------------------------------------------------
// Pointwise LSTM cell update. t >= 0: encoder mode, also record h_all/c_all
// (time-major [t][b][j], matching lax.scan output).
// ---------------------------------------------------------------------------
__device__ __forceinline__ float sigm(float x) {
  return 1.0f / (1.0f + __expf(-x));
}

__global__ void k_lstm_cell(const float* __restrict__ g,
                            float* __restrict__ h, float* __restrict__ c,
                            __bf16* __restrict__ hbf,
                            float* __restrict__ h_all,
                            float* __restrict__ c_all, int t) {
  const int idx = blockIdx.x * blockDim.x + threadIdx.x;   // B*H
  if (idx >= B_ * H_) return;
  const int b = idx >> 9, j = idx & (H_ - 1);
  const size_t gb = (size_t)b * G4H;
  const float ig = g[gb + j];
  const float fg = g[gb + H_ + j];
  const float gg = g[gb + 2 * H_ + j];
  const float og = g[gb + 3 * H_ + j];
  const float cn = sigm(fg) * c[idx] + sigm(ig) * tanhf(gg);
  const float hn = sigm(og) * tanhf(cn);
  c[idx] = cn;
  h[idx] = hn;
  hbf[idx] = (__bf16)hn;
  if (t >= 0) {
    h_all[(size_t)t * B_ * H_ + idx] = hn;
    c_all[(size_t)t * B_ * H_ + idx] = cn;
  }
}

__global__ void k_zero_state(float* __restrict__ h, float* __restrict__ c,
                             __bf16* __restrict__ hbf) {
  const int idx = blockIdx.x * blockDim.x + threadIdx.x;
  if (idx >= B_ * H_) return;
  h[idx] = 0.0f; c[idx] = 0.0f; hbf[idx] = (__bf16)0.0f;
}

// Gather decoder init state: h_init[b] = h_all[seq_m[b]-1, b]
__global__ void k_dec_init(const int* __restrict__ seq_m,
                           const float* __restrict__ h_all,
                           const float* __restrict__ c_all,
                           float* __restrict__ h, float* __restrict__ c,
                           __bf16* __restrict__ hbf) {
  const int idx = blockIdx.x * blockDim.x + threadIdx.x;
  if (idx >= B_ * H_) return;
  const int b = idx >> 9;
  const int len = seq_m[b] - 1;
  const float hv = h_all[(size_t)len * B_ * H_ + idx];
  const float cv = c_all[(size_t)len * B_ * H_ + idx];
  h[idx] = hv; c[idx] = cv; hbf[idx] = (__bf16)hv;
}

// e_bf16[b, l, :] = bf16(h_all[l, b, :])   (encoder outputs, batch-major)
__global__ void k_build_e(const float* __restrict__ h_all,
                          __bf16* __restrict__ ebf) {
  const int idx = blockIdx.x * blockDim.x + threadIdx.x;   // B*L*H
  if (idx >= B_ * L_ * H_) return;
  const int j = idx & (H_ - 1);
  const int l = (idx >> 9) & (L_ - 1);
  const int b = idx >> 16;                                  // / (L_*H_)
  ebf[idx] = (__bf16)h_all[(size_t)l * B_ * H_ + b * H_ + j];
}

__global__ void k_f32_to_bf16(const float* __restrict__ in,
                              __bf16* __restrict__ out, int n) {
  const int i = blockIdx.x * blockDim.x + threadIdx.x;
  if (i < n) out[i] = (__bf16)in[i];
}

// ---------------------------------------------------------------------------
// Attention: block per batch b, 128 threads (one per l).
// score = sum_h tanh(e[b,l,h] + d[b,h]) * v[h]; mask (seq_mod!=0); softmax;
// write p to out[b, step, :]; argmax -> d_next[b] = seq_mod[b, ptr].
// ---------------------------------------------------------------------------
__global__ void k_attn(const float* __restrict__ e,
                       const float* __restrict__ dvec,
                       const float* __restrict__ v,
                       const float* __restrict__ seq,
                       float* __restrict__ out_p,
                       float* __restrict__ xnext, int step) {
  const int b = blockIdx.x;
  const int l = threadIdx.x;
  __shared__ float red[L_];
  __shared__ int   redi[L_];

  const float* ebl = e + ((size_t)b * L_ + l) * H_;
  const float* db  = dvec + (size_t)b * H_;
  float s = 0.0f;
  for (int hh = 0; hh < H_; ++hh) s += tanhf(ebl[hh] + db[hh]) * v[hh];

  const float sv = (l == 0) ? 0.1f : seq[b * L_ + l];      // seq_mod
  const float logit = (sv != 0.0f) ? s : (s - 1000.0f);

  // max-reduce
  red[l] = logit; __syncthreads();
  for (int off = L_ / 2; off > 0; off >>= 1) {
    if (l < off) red[l] = fmaxf(red[l], red[l + off]);
    __syncthreads();
  }
  const float mx = red[0]; __syncthreads();

  const float ex = __expf(logit - mx);
  red[l] = ex; __syncthreads();
  for (int off = L_ / 2; off > 0; off >>= 1) {
    if (l < off) red[l] += red[l + off];
    __syncthreads();
  }
  const float sum = red[0]; __syncthreads();

  out_p[((size_t)b * T_ + step) * L_ + l] = ex / sum;

  // argmax (lowest index wins ties, matching jnp.argmax)
  red[l] = logit; redi[l] = l; __syncthreads();
  for (int off = L_ / 2; off > 0; off >>= 1) {
    if (l < off) {
      const float o = red[l + off]; const int oi = redi[l + off];
      if (o > red[l] || (o == red[l] && oi < redi[l])) { red[l] = o; redi[l] = oi; }
    }
    __syncthreads();
  }
  if (l == 0) {
    const int ptr = redi[0];
    xnext[b] = (ptr == 0) ? 0.1f : seq[b * L_ + ptr];
  }
}

// ---------------------------------------------------------------------------
extern "C" void kernel_launch(void* const* d_in, const int* in_sizes, int n_in,
                              void* d_out, int out_size, void* d_ws, size_t ws_size,
                              hipStream_t stream) {
  (void)in_sizes; (void)n_in; (void)out_size; (void)ws_size;
  const float* seq    = (const float*)d_in[0];
  const int*   seq_m  = (const int*)  d_in[1];
  const float* target = (const float*)d_in[2];
  const float* Wih_e  = (const float*)d_in[3];
  const float* Whh_e  = (const float*)d_in[4];
  const float* bih_e  = (const float*)d_in[5];
  const float* bhh_e  = (const float*)d_in[6];
  const float* Wih_d  = (const float*)d_in[7];
  const float* Whh_d  = (const float*)d_in[8];
  const float* bih_d  = (const float*)d_in[9];
  const float* bhh_d  = (const float*)d_in[10];
  const float* We     = (const float*)d_in[11];
  const float* be     = (const float*)d_in[12];
  const float* Wd     = (const float*)d_in[13];
  const float* bd     = (const float*)d_in[14];
  const float* v      = (const float*)d_in[15];
  float* out_p = (float*)d_out;

  // ---- workspace carve-up ----
  char* ws = (char*)d_ws;
  auto carve = [&](size_t bytes) -> void* {
    void* p = (void*)ws;
    ws += (bytes + 255) & ~(size_t)255;
    return p;
  };
  __bf16* whhE_bf = (__bf16*)carve((size_t)G4H * H_ * 2);
  __bf16* whhD_bf = (__bf16*)carve((size_t)G4H * H_ * 2);
  __bf16* we_bf   = (__bf16*)carve((size_t)H_ * H_ * 2);
  __bf16* wd_bf   = (__bf16*)carve((size_t)H_ * H_ * 2);
  float*  h_all   = (float*) carve((size_t)L_ * B_ * H_ * 4);
  float*  c_all   = (float*) carve((size_t)L_ * B_ * H_ * 4);
  float*  e_f32   = (float*) carve((size_t)B_ * L_ * H_ * 4);
  __bf16* e_bf0   = (__bf16*)carve((size_t)B_ * L_ * H_ * 2);
  __bf16* e_bf1   = (__bf16*)carve((size_t)B_ * L_ * H_ * 2);
  float*  gbuf    = (float*) carve((size_t)B_ * G4H * 4);
  float*  hbuf    = (float*) carve((size_t)B_ * H_ * 4);
  float*  cbuf    = (float*) carve((size_t)B_ * H_ * 4);
  __bf16* hbf     = (__bf16*)carve((size_t)B_ * H_ * 2);
  float*  dvec    = (float*) carve((size_t)B_ * H_ * 4);
  float*  xnext   = (float*) carve((size_t)B_ * 4);

  // ---- one-time weight conversion to bf16 ----
  k_f32_to_bf16<<<(G4H * H_ + 255) / 256, 256, 0, stream>>>(Whh_e, whhE_bf, G4H * H_);
  k_f32_to_bf16<<<(G4H * H_ + 255) / 256, 256, 0, stream>>>(Whh_d, whhD_bf, G4H * H_);
  k_f32_to_bf16<<<(H_ * H_ + 255) / 256, 256, 0, stream>>>(We, we_bf, H_ * H_);
  k_f32_to_bf16<<<(H_ * H_ + 255) / 256, 256, 0, stream>>>(Wd, wd_bf, H_ * H_);

  // ---- encoder ----
  const int stateBlocks = (B_ * H_ + 255) / 256;       // 128
  k_zero_state<<<stateBlocks, 256, 0, stream>>>(hbuf, cbuf, hbf);
  const int gateBlocks = ((B_ / 16) * (G4H / 16)) / 8; // 64 blocks x 8 waves
  for (int t = 0; t < L_; ++t) {
    k_lstm_gates<<<gateBlocks, 256, 0, stream>>>(hbf, whhE_bf, bih_e, bhh_e,
                                                 Wih_e, seq, target, xnext,
                                                 gbuf, /*mode=*/0, t);
    k_lstm_cell<<<stateBlocks, 256, 0, stream>>>(gbuf, hbuf, cbuf, hbf,
                                                 h_all, c_all, t);
  }

  // ---- decoder init ----
  k_dec_init<<<stateBlocks, 256, 0, stream>>>(seq_m, h_all, c_all,
                                              hbuf, cbuf, hbf);
  k_build_e<<<(B_ * L_ * H_ + 255) / 256, 256, 0, stream>>>(h_all, e_bf0);

  // ---- decoder loop ----
  const int dBlocks = ((B_ / 16) * (H_ / 16)) / 8;            // 16
  const int eBlocks = (((B_ * L_) / 16) * (H_ / 16)) / 8;     // 2048
  __bf16* e_cur = e_bf0;
  __bf16* e_nxt = e_bf1;
  for (int i = 0; i < T_; ++i) {
    k_lstm_gates<<<gateBlocks, 256, 0, stream>>>(hbf, whhD_bf, bih_d, bhh_d,
                                                 Wih_d, seq, target, xnext,
                                                 gbuf, /*mode=*/1, i);
    k_lstm_cell<<<stateBlocks, 256, 0, stream>>>(gbuf, hbuf, cbuf, hbf,
                                                 nullptr, nullptr, -1);
    // d = h @ Wd^T + bd
    k_gemm_bias<<<dBlocks, 256, 0, stream>>>(hbf, wd_bf, bd, dvec, nullptr,
                                             B_, H_, H_);
    // e <- e @ We^T + be   (f32 for attention + bf16 for next iteration)
    k_gemm_bias<<<eBlocks, 256, 0, stream>>>(e_cur, we_bf, be, e_f32, e_nxt,
                                             B_ * L_, H_, H_);
    k_attn<<<B_, L_, 0, stream>>>(e_f32, dvec, v, seq, out_p, xnext, i);
    __bf16* tmp = e_cur; e_cur = e_nxt; e_nxt = tmp;
  }
}
